// PrimaryNetwork_51316269252874
// MI455X (gfx1250) — matrix operand order repaired and checked by reading, not compiled, not run on hardware
//
#include <hip/hip_runtime.h>
#include <hip/hip_bf16.h>

typedef __attribute__((ext_vector_type(16))) __bf16 v16bf;
typedef __attribute__((ext_vector_type(8)))  float  v8f;
typedef __attribute__((ext_vector_type(4)))  unsigned int v4u;

#define OBS_N   64
#define ACT_N   32
#define H_N     512
#define HH_N    256
#define P_N     16
#define B_N     4096
#define MT      64          // batch rows per workgroup
#define THREADS 64          // 2 waves of 32; each wave owns 32 rows (2 subtiles)
#define W_TOTAL 846914

// flat f32 weight-buffer offsets (per ensemble member)
#define OFF_WOBS   0
#define OFF_BOBS   16384
#define OFF_WACT   16640
#define OFF_BACT   24832
#define OFF_LAYER0 25088
#define LAYER_SZ   (H_N*H_N + H_N)     // 262656
#define OFF_WSTATE 813056
#define OFF_BSTATE 845824
#define OFF_WREW   845888
#define OFF_BREW   846400
#define OFF_WDONE  846401
#define OFF_BDONE  846913

// swizzled bf16 workspace layout (halfs, per member)
// tile(ks,nt) = 32 lanes x 16 halfs contiguous (1KB); tiles ordered [ks][nt]
#define SW_WIN       0
#define SW_L0        (512*96)                   // 49152
#define SW_LAYER_LEN (512*512)                  // 262144
#define SW_WSTATE    (SW_L0 + 3*SW_LAYER_LEN)   // 835584
#define SW_PER_P     (SW_WSTATE + 64*512)       // 868352

#define CK 4                         // k-steps per staged chunk
#define CHUNK_HALFS (CK*4*512)       // 8192 halfs = 16 KB

// ---------------- CDNA5 async global->LDS copy path (signature probe-confirmed)
#if defined(__has_builtin)
# if __has_builtin(__builtin_amdgcn_global_load_async_to_lds_b128) && \
     __has_builtin(__builtin_amdgcn_s_wait_asynccnt)
#  define HAVE_ASYNC 1
# endif
#endif
#ifndef HAVE_ASYNC
# define HAVE_ASYNC 0
#endif

#if HAVE_ASYNC
typedef int v4i_raw __attribute__((vector_size(16)));
typedef __attribute__((address_space(1))) v4i_raw as1_v4i;
typedef __attribute__((address_space(3))) v4i_raw as3_v4i;
__device__ inline void async_cp16(const __bf16* gsrc, __bf16* ldst) {
  as1_v4i* g = (as1_v4i*)(unsigned long long)(size_t)gsrc;
  as3_v4i* l = (as3_v4i*)(unsigned int)(size_t)ldst;
  __builtin_amdgcn_global_load_async_to_lds_b128(g, l, 0, 0);
}
#endif

__device__ inline int imin(int a, int b) { return a < b ? a : b; }

// ---- A-matrix fragment (16x32 bf16) per CDNA5 ISA layout
__device__ inline v16bf load_a_frag(const __bf16* y, int stride, int m, int kbase, int hi) {
  v16bf a;
  unsigned int* ap = (unsigned int*)&a;
  const __bf16* Y = y + (size_t)m * stride + kbase + hi * 8;
#pragma unroll
  for (int v = 0; v < 8; ++v) {
    int K0 = (v < 4) ? (2 * v) : (16 + 2 * (v - 4));
    ap[v] = *(const unsigned int*)(Y + K0);
  }
  return a;
}

// ---- B-matrix fragment from pre-swizzled tile: lane's 32B contiguous
__device__ inline v16bf load_b_swz(const __bf16* tile, int l) {
  v16bf b;
  v4u* bp = (v4u*)&b;
  const v4u* s = (const v4u*)(tile + l * 16);
  bp[0] = s[0];
  bp[1] = s[1];
  return b;
}

// =====================================================================
// Kernel 1: convert + swizzle all GEMM weights f32 -> bf16 fragment order
// =====================================================================
__global__ __launch_bounds__(256)
void swizzle_weights(const float* __restrict__ w, __bf16* __restrict__ ws) {
  const size_t frags_per_p = SW_PER_P / 16;       // 54272
  size_t fid = (size_t)blockIdx.x * blockDim.x + threadIdx.x;
  if (fid >= (size_t)P_N * frags_per_p) return;
  int p = (int)(fid / frags_per_p);
  size_t off16 = (fid % frags_per_p) * 16;        // half offset within member
  const float* wp = w + (size_t)p * W_TOTAL;
  __bf16* op = ws + (size_t)p * SW_PER_P + off16;

  int mat, NT;
  size_t local;
  if (off16 < SW_L0)            { mat = 0; local = off16;               NT = 32; }
  else if (off16 < SW_WSTATE)   { mat = 1 + (int)((off16 - SW_L0) / SW_LAYER_LEN);
                                  local = (off16 - SW_L0) % SW_LAYER_LEN; NT = 32; }
  else                          { mat = 4; local = off16 - SW_WSTATE;   NT = 4; }

  int fl   = (int)(local / 16);
  int l    = fl & 31;
  int tile = fl >> 5;
  int nt   = tile % NT;
  int ks   = tile / NT;
  int lo = l & 15, hi = l >> 4;
  int n  = nt * 16 + lo;
  int kb = ks * 32 + hi * 8;
#pragma unroll
  for (int j = 0; j < 16; ++j) {
    int k = kb + (j < 8 ? j : 8 + j);
    float v;
    if (mat == 0) {             // virtual input weight 512 x 96 (block-diagonal)
      if (n < HH_N) v = (k < OBS_N) ? wp[OFF_WOBS + n * OBS_N + k] : 0.f;
      else          v = (k >= OBS_N) ? wp[OFF_WACT + (n - HH_N) * ACT_N + (k - OBS_N)] : 0.f;
    } else if (mat <= 3) {      // hidden layers 512 x 512
      v = wp[OFF_LAYER0 + (size_t)(mat - 1) * LAYER_SZ + (size_t)n * H_N + k];
    } else {                    // state head 64 x 512
      v = wp[OFF_WSTATE + (size_t)n * H_N + k];
    }
    op[j] = (__bf16)v;
  }
}

// ---- stage one chunk (<=4 ksteps x 4 n-tiles) of pre-swizzled weights
__device__ inline void copy_chunk(const __bf16* swz, int NTtot, int g, int ks0, int ksn,
                                  __bf16* dstbuf, int t) {
  for (int ksl = 0; ksl < ksn; ++ksl) {
    const __bf16* src = swz + ((size_t)(ks0 + ksl) * NTtot + g * 4) * 512;
    __bf16* dst = dstbuf + ksl * 2048;
#pragma unroll
    for (int u = 0; u < 4; ++u) {
      int e = t + u * THREADS;            // 0..255, 16B each -> 4KB per kstep
#if HAVE_ASYNC
      async_cp16(src + e * 8, dst + e * 8);
#else
      *(v4u*)(dst + e * 8) = *(const v4u*)(src + e * 8);
#endif
    }
  }
}

// ---- double-buffered GEMM accumulate: 2 M-subtiles x 4 N-tiles per wave,
//      B-fragments reused from registers across both M-subtiles.
__device__ inline void gemm_accum(const __bf16* swz_mat, int NTtot, int KS, int g,
                                  const __bf16* src, int sstride, __bf16* sWb,
                                  int t, int mbase, int lo, int hi, int l,
                                  v8f acc0[4], v8f acc1[4]) {
#pragma unroll
  for (int nt = 0; nt < 4; ++nt) {
    acc0[nt] = (v8f){0, 0, 0, 0, 0, 0, 0, 0};
    acc1[nt] = (v8f){0, 0, 0, 0, 0, 0, 0, 0};
  }
  const int nch = (KS + CK - 1) / CK;
  copy_chunk(swz_mat, NTtot, g, 0, imin(CK, KS), sWb, t);
  for (int c = 0; c < nch; ++c) {
    __bf16* cur = sWb + (c & 1) * CHUNK_HALFS;
    const bool more = (c + 1 < nch);
    if (more)
      copy_chunk(swz_mat, NTtot, g, (c + 1) * CK, imin(CK, KS - (c + 1) * CK),
                 sWb + ((c + 1) & 1) * CHUNK_HALFS, t);
#if HAVE_ASYNC
    if (more) __builtin_amdgcn_s_wait_asynccnt(16);  // chunk c done, c+1 may fly
    else      __builtin_amdgcn_s_wait_asynccnt(0);
#endif
    __syncthreads();                                  // all waves' chunk c visible
    const int ksn = imin(CK, KS - c * CK);
    for (int ksl = 0; ksl < ksn; ++ksl) {
      const int kb = (c * CK + ksl) * 32;
      v16bf a0 = load_a_frag(src, sstride, mbase + lo,      kb, hi);
      v16bf a1 = load_a_frag(src, sstride, mbase + 16 + lo, kb, hi);
#pragma unroll
      for (int nt = 0; nt < 4; ++nt) {
        v16bf bb = load_b_swz(cur + (ksl * 4 + nt) * 512, l);
        acc0[nt] = __builtin_amdgcn_wmma_f32_16x16x32_bf16(false, a0, false, bb,
                                                           (short)0, acc0[nt], false, false);
        acc1[nt] = __builtin_amdgcn_wmma_f32_16x16x32_bf16(false, a1, false, bb,
                                                           (short)0, acc1[nt], false, false);
      }
    }
    __syncthreads();                                  // buffer reusable next+1 chunk
  }
}

// =====================================================================
// Kernel 2: fused ensemble MLP, activations resident in LDS
// =====================================================================
__global__ __launch_bounds__(THREADS)
void ensemble_mlp_kernel(const float* __restrict__ obs,
                         const float* __restrict__ action,
                         const float* __restrict__ w,
                         const float* __restrict__ scal,
                         const __bf16* __restrict__ wsw,
                         float* __restrict__ out_states,
                         float* __restrict__ out_rew,
                         float* __restrict__ out_done) {
  __shared__ __bf16 sY0[MT * H_N];        // 64 KB
  __shared__ __bf16 sY1[MT * H_N];        // 64 KB
  __shared__ __bf16 sWb[2 * CHUNK_HALFS]; // 32 KB double-buffered weight slab

  const int t     = threadIdx.x;
  const int wave  = t >> 5;
  const int lane  = t & 31;
  const int lo    = lane & 15;
  const int hi    = lane >> 4;
  const int mbase = wave * 32;            // each wave owns 32 rows
  const int mtile = blockIdx.x;
  const int p     = blockIdx.y;
  const int bbase = mtile * MT;
  const float*  wp  = w + (size_t)p * W_TOTAL;
  const __bf16* wsp = wsw + (size_t)p * SW_PER_P;

  // ---------------- stage scaled inputs x = [obs*scal, action] into sY1 (stride 96)
  for (int idx = t; idx < MT * OBS_N; idx += THREADS) {
    int m = idx >> 6, o = idx & 63;
    sY1[m * 96 + o] = (__bf16)(obs[(size_t)(bbase + m) * OBS_N + o] * scal[o]);
  }
  for (int idx = t; idx < MT * ACT_N; idx += THREADS) {
    int m = idx >> 5, a2 = idx & 31;
    sY1[m * 96 + 64 + a2] = (__bf16)(action[(size_t)(bbase + m) * ACT_N + a2]);
  }
  __syncthreads();

  // ---------------- input layer: x(64x96) @ Win^T -> sY0 (leaky relu)
  for (int g = 0; g < 8; ++g) {
    v8f acc0[4], acc1[4];
    gemm_accum(wsp + SW_WIN, 32, 3, g, sY1, 96, sWb, t, mbase, lo, hi, lane, acc0, acc1);
#pragma unroll
    for (int nt = 0; nt < 4; ++nt) {
      int n = g * 64 + nt * 16 + lo;
      float bias = (n < HH_N) ? wp[OFF_BOBS + n] : wp[OFF_BACT + (n - HH_N)];
#pragma unroll
      for (int r = 0; r < 8; ++r) {
        int m0 = mbase + hi * 8 + r;
        float v0 = acc0[nt][r] + bias;
        v0 = v0 > 0.f ? v0 : 0.01f * v0;
        sY0[m0 * H_N + n] = (__bf16)v0;
        float v1 = acc1[nt][r] + bias;
        v1 = v1 > 0.f ? v1 : 0.01f * v1;
        sY0[(m0 + 16) * H_N + n] = (__bf16)v1;
      }
    }
  }
  __syncthreads();

  // ---------------- 3 hidden layers, ping-pong sY0 <-> sY1
  const __bf16* src = sY0;
  __bf16* dst = sY1;
  for (int layer = 0; layer < 3; ++layer) {
    const __bf16* wmat = wsp + SW_L0 + (size_t)layer * SW_LAYER_LEN;
    const float*  bvec = wp + OFF_LAYER0 + (size_t)layer * LAYER_SZ + H_N * H_N;
    for (int g = 0; g < 8; ++g) {
      v8f acc0[4], acc1[4];
      gemm_accum(wmat, 32, 16, g, src, H_N, sWb, t, mbase, lo, hi, lane, acc0, acc1);
#pragma unroll
      for (int nt = 0; nt < 4; ++nt) {
        int n = g * 64 + nt * 16 + lo;
        float bias = bvec[n];
#pragma unroll
        for (int r = 0; r < 8; ++r) {
          int m0 = mbase + hi * 8 + r;
          float v0 = acc0[nt][r] + bias;
          v0 = v0 > 0.f ? v0 : 0.01f * v0;
          dst[m0 * H_N + n] = (__bf16)v0;
          float v1 = acc1[nt][r] + bias;
          v1 = v1 > 0.f ? v1 : 0.01f * v1;
          dst[(m0 + 16) * H_N + n] = (__bf16)v1;
        }
      }
    }
    __syncthreads();
    const __bf16* tmp = src; src = dst; dst = (__bf16*)tmp;
  }
  // src now holds the final hidden activation

  // ---------------- state head + residual: out = (gemm + b)/scal + obs
  {
    v8f acc0[4], acc1[4];
    gemm_accum(wsp + SW_WSTATE, 4, 16, 0, src, H_N, sWb, t, mbase, lo, hi, lane, acc0, acc1);
#pragma unroll
    for (int nt = 0; nt < 4; ++nt) {
      int n = nt * 16 + lo;                 // 0..63
      float bias  = wp[OFF_BSTATE + n];
      float inv_s = 1.0f / scal[n];
#pragma unroll
      for (int r = 0; r < 8; ++r) {
        int m0 = mbase + hi * 8 + r;
        {
          int bglob = bbase + m0;
          float v = (acc0[nt][r] + bias) * inv_s + obs[(size_t)bglob * OBS_N + n];
          out_states[((size_t)bglob * P_N + p) * OBS_N + n] = v;
        }
        {
          int bglob = bbase + m0 + 16;
          float v = (acc1[nt][r] + bias) * inv_s + obs[(size_t)bglob * OBS_N + n];
          out_states[((size_t)bglob * P_N + p) * OBS_N + n] = v;
        }
      }
    }
  }

  // ---------------- reward / done heads: 64 rows x 2 outputs == 128 tasks
  for (int task = t; task < 2 * MT; task += THREADS) {
    int m     = task >> 1;
    int which = task & 1;
    const float* wvec = wp + (which ? OFF_WDONE : OFF_WREW);
    float bias        = wp[which ? OFF_BDONE : OFF_BREW];
    float s = 0.f;
#pragma unroll 8
    for (int k = 0; k < H_N; ++k) s += (float)src[m * H_N + k] * wvec[k];
    s += bias;
    size_t o = (size_t)(bbase + m) * P_N + p;
    if (which) out_done[o] = 1.0f / (1.0f + __expf(-s));
    else       out_rew[o]  = s;
  }
}

extern "C" void kernel_launch(void* const* d_in, const int* in_sizes, int n_in,
                              void* d_out, int out_size, void* d_ws, size_t ws_size,
                              hipStream_t stream) {
  (void)in_sizes; (void)n_in; (void)ws_size; (void)out_size;
  const float* obs    = (const float*)d_in[0];
  const float* action = (const float*)d_in[1];
  const float* w      = (const float*)d_in[2];
  const float* scal   = (const float*)d_in[3];
  __bf16* wsw = (__bf16*)d_ws;                               // 27.8 MB swizzled bf16
  float* out        = (float*)d_out;
  float* out_states = out;                                   // B*P*OBS
  float* out_rew    = out + (size_t)B_N * P_N * OBS_N;       // B*P
  float* out_done   = out_rew + (size_t)B_N * P_N;           // B*P

  {
    size_t frags = (size_t)P_N * (SW_PER_P / 16);
    int blocks = (int)((frags + 255) / 256);
    hipLaunchKernelGGL(swizzle_weights, dim3(blocks), dim3(256), 0, stream, w, wsw);
  }
  dim3 grid(B_N / MT, P_N);
  hipLaunchKernelGGL(ensemble_mlp_kernel, grid, dim3(THREADS), 0, stream,
                     obs, action, w, scal, wsw, out_states, out_rew, out_done);
}
